// QuantumLayer_13709535609620
// MI455X (gfx1250) — matrix operand order repaired and checked by reading, not compiled, not run on hardware
//
#include <hip/hip_runtime.h>
#include <stdint.h>

// ---------------------------------------------------------------------------
// QuantumLayer for MI455X (gfx1250), compile-only target.
//
// Math: per split s, expvals are linear in the 81 products of {1,cos,sin} of
// the 4 angles (rr^T = tensor of 0.5(I + cos*Z + sin*X)). We fold the
// per-split 12x81 constant maps (from simulating the fixed 16x16 unitary) and
// mw into W1, giving one dense f16 WMMA GEMM 16384x1296 -> 256, then two more
// WMMA GEMMs for the MLP tail. f32 accumulation everywhere.
// ---------------------------------------------------------------------------

typedef _Float16 half_t;
typedef __attribute__((ext_vector_type(16))) _Float16 v16h;
typedef __attribute__((ext_vector_type(8)))  _Float16 v8h;
typedef __attribute__((ext_vector_type(8)))  float    v8f;

#define N_ROWS   16384   // B*S = 8*2048
#define NSPLIT   16
#define KPHI     1296    // 16 splits * 81 basis fns
#define KPAD     1312    // 41 * 32 (WMMA K step)
#define D1       256
#define D2       128
#define D3       64

// workspace offsets (bytes), all 256B aligned
#define OFF_C      0u         // 16*972 f32      = 62,208 B
#define OFF_WFOLD  65536u     // 256*1312 f16    = 671,744 B
#define OFF_W2H    737280u    // 128*256 f16     = 65,536 B
#define OFF_W3H    802816u    // 64*128 f16      = 16,384 B
#define OFF_PHI    819200u    // 16384*1312 f16  = 42,991,616 B
#define OFF_H1     43810816u  // 16384*256 f16   = 8,388,608 B
#define OFF_H2     52199424u  // 16384*128 f16   = 4,194,304 B

// ---------------------------------------------------------------------------
// Setup: simulate the fixed 4-qubit circuit per split, derive 12x81 coeffs.
// ---------------------------------------------------------------------------
struct C2x2 { float rr[2][2]; float ii[2][2]; };

__device__ __forceinline__ C2x2 mk_rx(float th) {
  float s, c; __sincosf(0.5f * th, &s, &c);
  C2x2 g{}; g.rr[0][0] = c; g.rr[1][1] = c; g.ii[0][1] = -s; g.ii[1][0] = -s; return g;
}
__device__ __forceinline__ C2x2 mk_ry(float th) {
  float s, c; __sincosf(0.5f * th, &s, &c);
  C2x2 g{}; g.rr[0][0] = c; g.rr[0][1] = -s; g.rr[1][0] = s; g.rr[1][1] = c; return g;
}
__device__ __forceinline__ C2x2 mk_rz(float th) {
  float s, c; __sincosf(0.5f * th, &s, &c);
  C2x2 g{}; g.rr[0][0] = c; g.ii[0][0] = -s; g.rr[1][1] = c; g.ii[1][1] = s; return g;
}
__device__ __forceinline__ C2x2 cmul2(const C2x2& A, const C2x2& B) {
  C2x2 r{};
#pragma unroll
  for (int i = 0; i < 2; i++)
#pragma unroll
    for (int j = 0; j < 2; j++) {
      float rr = 0.f, ri = 0.f;
#pragma unroll
      for (int k = 0; k < 2; k++) {
        rr += A.rr[i][k] * B.rr[k][j] - A.ii[i][k] * B.ii[k][j];
        ri += A.rr[i][k] * B.ii[k][j] + A.ii[i][k] * B.rr[k][j];
      }
      r.rr[i][j] = rr; r.ii[i][j] = ri;
    }
  return r;
}
// apply 2x2 complex gate on qubit with bit-mask m (qubit q -> m = 8>>q)
__device__ __forceinline__ void apply1q(float* vr, float* vi, int m, const C2x2& g) {
#pragma unroll
  for (int p = 0; p < 16; p++)
    if (!(p & m)) {
      int p1 = p | m;
      float ar = vr[p], ai = vi[p], br = vr[p1], bi = vi[p1];
      vr[p]  = g.rr[0][0]*ar - g.ii[0][0]*ai + g.rr[0][1]*br - g.ii[0][1]*bi;
      vi[p]  = g.rr[0][0]*ai + g.ii[0][0]*ar + g.rr[0][1]*bi + g.ii[0][1]*br;
      vr[p1] = g.rr[1][0]*ar - g.ii[1][0]*ai + g.rr[1][1]*br - g.ii[1][1]*bi;
      vi[p1] = g.rr[1][0]*ai + g.ii[1][0]*ar + g.rr[1][1]*bi + g.ii[1][1]*br;
    }
}
__device__ __forceinline__ void applyCtrl(float* vr, float* vi, int cm, int tm, const C2x2& g) {
#pragma unroll
  for (int p = 0; p < 16; p++)
    if ((p & cm) && !(p & tm)) {
      int p1 = p | tm;
      float ar = vr[p], ai = vi[p], br = vr[p1], bi = vi[p1];
      vr[p]  = g.rr[0][0]*ar - g.ii[0][0]*ai + g.rr[0][1]*br - g.ii[0][1]*bi;
      vi[p]  = g.rr[0][0]*ai + g.ii[0][0]*ar + g.rr[0][1]*bi + g.ii[0][1]*br;
      vr[p1] = g.rr[1][0]*ar - g.ii[1][0]*ai + g.rr[1][1]*br - g.ii[1][1]*bi;
      vi[p1] = g.rr[1][0]*ai + g.ii[1][0]*ar + g.rr[1][1]*bi + g.ii[1][1]*br;
    }
}

// qw, tw: [split][layer][qubit] = s*12 + l*4 + q ; mw: [split][3]
// C out: [s][k][i] with k = P*4 + q (Z,X,Y groups), i = base-3 index over
// (i0,i1,i2,i3), basis {1, cos, sin} per qubit. C[s*972 + k*81 + i].
__global__ __launch_bounds__(256) void setup_coeff(
    const float* __restrict__ qw, const float* __restrict__ tw,
    const float* __restrict__ mw, float* __restrict__ C) {
  const int s = blockIdx.x;
  const int t = threadIdx.x;
  __shared__ float Ur[16][16], Ui[16][16];
  __shared__ float Smat[12][16][16];

  // ---- Phase 1: 16 threads evolve basis columns of U through the circuit
  if (t < 16) {
    float vr[16], vi[16];
#pragma unroll
    for (int p = 0; p < 16; p++) { vr[p] = (p == t) ? 1.f : 0.f; vi[p] = 0.f; }
#pragma unroll
    for (int l = 0; l < 3; l++) {
#pragma unroll
      for (int q = 0; q < 4; q++) {
        float th = qw[s * 12 + l * 4 + q];
        C2x2 U1 = cmul2(mk_rz(th), cmul2(mk_ry(th), mk_rx(th)));  // RZ*RY*RX
        apply1q(vr, vi, 8 >> q, U1);
      }
#pragma unroll
      for (int i = 0; i < 3; i++) {
        int cm = 8 >> i, tm = 8 >> (i + 1);
        applyCtrl(vr, vi, cm, tm, mk_rx(tw[s * 12 + l * 4 + i]));
        // CX: swap within controlled pairs
#pragma unroll
        for (int p = 0; p < 16; p++)
          if ((p & cm) && !(p & tm)) {
            int p1 = p | tm;
            float tr = vr[p]; vr[p] = vr[p1]; vr[p1] = tr;
            float ti = vi[p]; vi[p] = vi[p1]; vi[p1] = ti;
          }
        // CZ: negate control&target set
#pragma unroll
        for (int p = 0; p < 16; p++)
          if ((p & cm) && (p & tm)) { vr[p] = -vr[p]; vi[p] = -vi[p]; }
      }
      {  // ring closure: control q3 (mask 1), target q0 (mask 8)
        int cm = 1, tm = 8;
        applyCtrl(vr, vi, cm, tm, mk_rx(tw[s * 12 + l * 4 + 3]));
#pragma unroll
        for (int p = 0; p < 16; p++)
          if ((p & cm) && !(p & tm)) {
            int p1 = p | tm;
            float tr = vr[p]; vr[p] = vr[p1]; vr[p1] = tr;
            float ti = vi[p]; vi[p] = vi[p1]; vi[p1] = ti;
          }
#pragma unroll
        for (int p = 0; p < 16; p++)
          if ((p & cm) && (p & tm)) { vr[p] = -vr[p]; vi[p] = -vi[p]; }
      }
      // SWAP qubits 0,1 : exchange bits 3 and 2
#pragma unroll
      for (int p = 0; p < 16; p++)
        if (((p >> 3) & 1) == 1 && ((p >> 2) & 1) == 0) {
          int pp = p ^ 12;
          float tr = vr[p]; vr[p] = vr[pp]; vr[pp] = tr;
          float ti = vi[p]; vi[p] = vi[pp]; vi[pp] = ti;
        }
    }
#pragma unroll
    for (int p = 0; p < 16; p++) { Ur[p][t] = vr[p]; Ui[p][t] = vi[p]; }
  }
  __syncthreads();

  // ---- Phase 2: S_k[p][q] = Re(i^{pc(p)} (-i)^{pc(q)} (U^dag P_k U)[p][q])
  {
    int p = t >> 4, q = t & 15;
#pragma unroll
    for (int k = 0; k < 12; k++) {
      int P = k >> 2, w = k & 3, m = 8 >> w;
      float mr = 0.f, mi = 0.f;
#pragma unroll
      for (int a = 0; a < 16; a++) {
        float ur = Ur[a][p], ui = Ui[a][p];  // conj(U[a][p]) = (ur, -ui)
        if (P == 0) {  // Z
          float z = (a & m) ? -1.f : 1.f;
          float br = Ur[a][q], bi = Ui[a][q];
          mr += z * (ur * br + ui * bi);
          mi += z * (ur * bi - ui * br);
        } else {
          int b = a ^ m;
          float br = Ur[b][q], bi = Ui[b][q];
          float pr = ur * br + ui * bi, pi = ur * bi - ui * br;
          if (P == 1) { mr += pr; mi += pi; }          // X
          else if (a & m) { mr -= pi; mi += pr; }      // Y, coef +i
          else            { mr += pi; mi -= pr; }      // Y, coef -i
        }
      }
      int d = (__popc(p) - __popc(q)) & 3;
      float sv = (d == 0) ? mr : (d == 1) ? -mi : (d == 2) ? -mr : mi;
      Smat[k][p][q] = sv;
    }
  }
  __syncthreads();

  // ---- Phase 3: coeff[k][i] = mw/16 * sum_p sign(p&zmask) * S[p][p^xmask]
  for (int out = t; out < 12 * 81; out += 256) {
    int k = out / 81, i = out % 81;
    int i0 = i / 27, i1 = (i / 9) % 3, i2 = (i / 3) % 3, i3 = i % 3;
    int zmask = (i0 == 1 ? 8 : 0) | (i1 == 1 ? 4 : 0) | (i2 == 1 ? 2 : 0) | (i3 == 1 ? 1 : 0);
    int xmask = (i0 == 2 ? 8 : 0) | (i1 == 2 ? 4 : 0) | (i2 == 2 ? 2 : 0) | (i3 == 2 ? 1 : 0);
    float acc = 0.f;
#pragma unroll
    for (int p = 0; p < 16; p++) {
      float sg = (__popc(p & zmask) & 1) ? -1.f : 1.f;
      acc += sg * Smat[k][p][p ^ xmask];
    }
    C[s * 972 + out] = acc * 0.0625f * mw[s * 3 + (k >> 2)];
  }
}

// Fold coeffs into W1: Wfold[n][s*81+i] = sum_k W1[n][s*12+k]*C[s][k][i], f16
__global__ __launch_bounds__(256) void setup_fold(
    const float* __restrict__ W1, const float* __restrict__ C,
    half_t* __restrict__ Wfold) {
  int n = blockIdx.x;
  for (int col = threadIdx.x; col < KPAD; col += 256) {
    float v = 0.f;
    if (col < KPHI) {
      int s = col / 81, i = col % 81;
#pragma unroll
      for (int k = 0; k < 12; k++)
        v += W1[n * 192 + s * 12 + k] * C[s * 972 + k * 81 + i];
    }
    Wfold[n * KPAD + col] = (half_t)v;
  }
}

__global__ __launch_bounds__(256) void conv_w(
    const float* __restrict__ W2, const float* __restrict__ W3,
    half_t* __restrict__ W2h, half_t* __restrict__ W3h) {
  int i = blockIdx.x * 256 + threadIdx.x;
  if (i < D2 * D1) W2h[i] = (half_t)W2[i];
  int j = i - D2 * D1;
  if (j >= 0 && j < D3 * D2) W3h[j] = (half_t)W3[j];
}

// ---------------------------------------------------------------------------
// Phi features: per (row, split) the 81 products of {1, cos x_q, sin x_q}
// ---------------------------------------------------------------------------
__global__ __launch_bounds__(256) void phi_kernel(
    const float* __restrict__ x, half_t* __restrict__ Phi) {
  int idx = blockIdx.x * 256 + threadIdx.x;
  int n = idx >> 4, s = idx & 15;
  const float* xp = x + n * 64 + s * 4;
  float cs[4], sn[4];
#pragma unroll
  for (int q = 0; q < 4; q++) __sincosf(xp[q], &sn[q], &cs[q]);
  float t0[3] = {1.f, cs[0], sn[0]}, t1[3] = {1.f, cs[1], sn[1]};
  float t2[3] = {1.f, cs[2], sn[2]}, t3[3] = {1.f, cs[3], sn[3]};
  float p01[9], p23[9];
#pragma unroll
  for (int a = 0; a < 3; a++)
#pragma unroll
    for (int b = 0; b < 3; b++) {
      p01[a * 3 + b] = t0[a] * t1[b];
      p23[a * 3 + b] = t2[a] * t3[b];
    }
  half_t* outp = Phi + (size_t)n * KPAD + s * 81;
#pragma unroll
  for (int a = 0; a < 9; a++)
#pragma unroll
    for (int b = 0; b < 9; b++) outp[a * 9 + b] = (half_t)(p01[a] * p23[b]);
  if (s == 15) {
    half_t* pad = Phi + (size_t)n * KPAD + KPHI;
#pragma unroll
    for (int j = 0; j < 16; j++) pad[j] = (half_t)0.f;
  }
}

// ---------------------------------------------------------------------------
// WMMA fragment loaders (wave32; this code's contiguous K order, same for A and B)
// A (16x32 f16): lane<16 row=lane, halves = K{k0..k0+15};
//                lane>=16 row=lane-16, K{k0+16..k0+31}
// B: per lane 16 K-contiguous halves; operand stored row-major [n][K].
// ---------------------------------------------------------------------------
__device__ __forceinline__ v16h load_afrag(const half_t* row, int k0, int hi) {
  int ka = k0 + hi * 16;
  v8h a0 = *(const v8h*)(row + ka);
  v8h a1 = *(const v8h*)(row + ka + 8);
  return __builtin_shufflevector(a0, a1, 0, 1, 2, 3, 4, 5, 6, 7,
                                 8, 9, 10, 11, 12, 13, 14, 15);
}

// GEMM1: H1[16384,256] = leaky(Phi[16384,KPAD] * Wfold^T + b1), f16 out
__global__ __launch_bounds__(256) void gemm1(
    const half_t* __restrict__ Phi, const half_t* __restrict__ Wfold,
    const float* __restrict__ bias1, half_t* __restrict__ H1) {
  const int lane = threadIdx.x & 31, wv = threadIdx.x >> 5;
  const int rowBase = blockIdx.x * 16, colBase = wv * 32;
  const int lrow = lane & 15, hi = lane >> 4;
  v8f acc0 = {}, acc1 = {};
  const half_t* arow  = Phi + (size_t)(rowBase + lrow) * KPAD;
  const half_t* brow0 = Wfold + (size_t)(colBase + lrow) * KPAD;
  const half_t* brow1 = Wfold + (size_t)(colBase + 16 + lrow) * KPAD;
  for (int k0 = 0; k0 < KPAD; k0 += 32) {
    __builtin_prefetch(arow + k0 + 64, 0, 1);
    v16h a  = load_afrag(arow, k0, hi);
    int kb  = k0 + hi * 16;
    v16h b0 = *(const v16h*)(brow0 + kb);
    v16h b1 = *(const v16h*)(brow1 + kb);
    acc0 = __builtin_amdgcn_wmma_f32_16x16x32_f16(false, a, false, b0, (short)0, acc0, false, false);
    acc1 = __builtin_amdgcn_wmma_f32_16x16x32_f16(false, a, false, b1, (short)0, acc1, false, false);
  }
  int col0 = colBase + lrow, col1 = colBase + 16 + lrow;
  float bb0 = bias1[col0], bb1 = bias1[col1];
#pragma unroll
  for (int v = 0; v < 8; v++) {
    int m = v + hi * 8;
    float x0 = acc0[v] + bb0; x0 = x0 > 0.f ? x0 : 0.1f * x0;
    float x1 = acc1[v] + bb1; x1 = x1 > 0.f ? x1 : 0.1f * x1;
    H1[(size_t)(rowBase + m) * D1 + col0] = (half_t)x0;
    H1[(size_t)(rowBase + m) * D1 + col1] = (half_t)x1;
  }
}

// GEMM2: H2[16384,128] = leaky(H1 * W2h^T + b2)
__global__ __launch_bounds__(256) void gemm2(
    const half_t* __restrict__ H1, const half_t* __restrict__ W2h,
    const float* __restrict__ bias2, half_t* __restrict__ H2) {
  const int lane = threadIdx.x & 31, wv = threadIdx.x >> 5;
  const int rowBase = blockIdx.x * 16, colBase = wv * 16;
  const int lrow = lane & 15, hi = lane >> 4;
  v8f acc = {};
  const half_t* arow = H1 + (size_t)(rowBase + lrow) * D1;
  const half_t* brow = W2h + (size_t)(colBase + lrow) * D1;
#pragma unroll
  for (int k0 = 0; k0 < D1; k0 += 32) {
    v16h a = load_afrag(arow, k0, hi);
    v16h b = *(const v16h*)(brow + k0 + hi * 16);
    acc = __builtin_amdgcn_wmma_f32_16x16x32_f16(false, a, false, b, (short)0, acc, false, false);
  }
  int col = colBase + lrow;
  float bb = bias2[col];
#pragma unroll
  for (int v = 0; v < 8; v++) {
    int m = v + hi * 8;
    float x0 = acc[v] + bb; x0 = x0 > 0.f ? x0 : 0.1f * x0;
    H2[(size_t)(rowBase + m) * D2 + col] = (half_t)x0;
  }
}

// GEMM3: out[16384,64] = relu(H2 * W3h^T + b3), f32 out
__global__ __launch_bounds__(128) void gemm3(
    const half_t* __restrict__ H2, const half_t* __restrict__ W3h,
    const float* __restrict__ bias3, float* __restrict__ outp) {
  const int lane = threadIdx.x & 31, wv = threadIdx.x >> 5;  // wv 0..3
  const int rowBase = blockIdx.x * 16, colBase = wv * 16;
  const int lrow = lane & 15, hi = lane >> 4;
  v8f acc = {};
  const half_t* arow = H2 + (size_t)(rowBase + lrow) * D2;
  const half_t* brow = W3h + (size_t)(colBase + lrow) * D2;
#pragma unroll
  for (int k0 = 0; k0 < D2; k0 += 32) {
    v16h a = load_afrag(arow, k0, hi);
    v16h b = *(const v16h*)(brow + k0 + hi * 16);
    acc = __builtin_amdgcn_wmma_f32_16x16x32_f16(false, a, false, b, (short)0, acc, false, false);
  }
  int col = colBase + lrow;
  float bb = bias3[col];
#pragma unroll
  for (int v = 0; v < 8; v++) {
    int m = v + hi * 8;
    float x0 = acc[v] + bb;
    outp[(size_t)(rowBase + m) * D3 + col] = x0 > 0.f ? x0 : 0.f;
  }
}

// ---------------------------------------------------------------------------
extern "C" void kernel_launch(void* const* d_in, const int* in_sizes, int n_in,
                              void* d_out, int out_size, void* d_ws, size_t ws_size,
                              hipStream_t stream) {
  (void)in_sizes; (void)n_in; (void)out_size; (void)ws_size;
  const float* x  = (const float*)d_in[0];
  const float* qw = (const float*)d_in[1];
  const float* tw = (const float*)d_in[2];
  const float* mw = (const float*)d_in[3];
  const float* W1 = (const float*)d_in[4];
  const float* b1 = (const float*)d_in[5];
  const float* W2 = (const float*)d_in[6];
  const float* b2 = (const float*)d_in[7];
  const float* W3 = (const float*)d_in[8];
  const float* b3 = (const float*)d_in[9];
  float* outp = (float*)d_out;

  uint8_t* ws = (uint8_t*)d_ws;
  float*  C     = (float*)(ws + OFF_C);
  half_t* Wfold = (half_t*)(ws + OFF_WFOLD);
  half_t* W2h   = (half_t*)(ws + OFF_W2H);
  half_t* W3h   = (half_t*)(ws + OFF_W3H);
  half_t* Phi   = (half_t*)(ws + OFF_PHI);
  half_t* H1    = (half_t*)(ws + OFF_H1);
  half_t* H2    = (half_t*)(ws + OFF_H2);

  setup_coeff<<<dim3(NSPLIT), dim3(256), 0, stream>>>(qw, tw, mw, C);
  setup_fold<<<dim3(D1), dim3(256), 0, stream>>>(W1, C, Wfold);
  conv_w<<<dim3((D2 * D1 + D3 * D2) / 256), dim3(256), 0, stream>>>(W2, W3, W2h, W3h);
  phi_kernel<<<dim3(N_ROWS * NSPLIT / 256), dim3(256), 0, stream>>>(x, Phi);
  gemm1<<<dim3(N_ROWS / 16), dim3(256), 0, stream>>>(Phi, Wfold, b1, H1);
  gemm2<<<dim3(N_ROWS / 16), dim3(256), 0, stream>>>(H1, W2h, b2, H2);
  gemm3<<<dim3(N_ROWS / 16), dim3(128), 0, stream>>>(H2, W3h, b3, outp);
}
